// TransformerForBiaffineParsing_9543417332214
// MI455X (gfx1250) — compile-verified, matrix operand order checked by reading, not run on hardware
//
#include <hip/hip_runtime.h>
#include <hip/hip_bf16.h>
#include <math.h>

typedef __bf16 bf16;
typedef __attribute__((ext_vector_type(16))) __bf16 v16bf;
typedef __attribute__((ext_vector_type(8)))  float  v8f;

#define B_   16
#define S_   256
#define H_   768
#define L_   37
#define Y_   512      // 2S
#define HP1  769      // H+1
#define JPAD 776      // padded T row length (bf16), multiple of 8 -> 16B aligned rows
#define LDSS 40       // LDS tile row stride in bf16 elems (32 + 8 pad)
#define TILE_ELEMS (64 * LDSS)

// ---------------------------------------------------------------------------
// Wait for this wave's outstanding async (global<->LDS) ops.
// ---------------------------------------------------------------------------
__device__ __forceinline__ void wait_async0() {
#if __has_builtin(__builtin_amdgcn_s_wait_asynccnt)
  __builtin_amdgcn_s_wait_asynccnt(0);
#else
  asm volatile("s_wait_asynccnt 0x0" ::: "memory");
#endif
}

// ---------------------------------------------------------------------------
// Async tile stage: 64 rows x 32 K-cols of bf16, global -> LDS directly via
// GLOBAL_LOAD_ASYNC_TO_LDS_B128 (ASYNCcnt), no VGPR round-trip. OOB rows are
// zero-filled with plain DS stores (published by the next __syncthreads).
// The instruction adds INST_OFFSET to both the LDS and global addresses, so
// the second 16B chunk is just "offset:16". blockDim.x must be 128.
// ---------------------------------------------------------------------------
__device__ __forceinline__ void issue_tile_async(const bf16* __restrict__ src, long ldRow,
                                                 int rowBase, int rowMax, int k0,
                                                 bf16* __restrict__ dst) {
  int t   = threadIdx.x;        // 0..127
  int row = t >> 1;             // 0..63
  int c0  = (t & 1) << 4;       // 0 or 16
  int gr  = rowBase + row;
  bf16* d = dst + row * LDSS + c0;
  if (gr < rowMax) {
    const bf16* s = src + (long)gr * ldRow + k0 + c0;
    unsigned lds_off = (unsigned)(size_t)d;   // LDS aperture: addr[31:0] = LDS offset
    unsigned long long ga = (unsigned long long)(size_t)s;
    asm volatile("global_load_async_to_lds_b128 %0, %1, off\n\t"
                 "global_load_async_to_lds_b128 %0, %1, off offset:16"
                 :: "v"(lds_off), "v"(ga) : "memory");
  } else {
    uint4 z; z.x = z.y = z.z = z.w = 0u;
    ((uint4*)d)[0] = z;
    ((uint4*)d)[1] = z;
  }
}

// ---------------------------------------------------------------------------
// Fragment load matching CDNA5 16-bit A/B layout for 16x16x32:
// lane l holds row (l&15); K elems: k = (e/8)*16 + ((l>=16)?8:0) + (e%8)
// -> two contiguous 16B LDS reads per fragment.
// ---------------------------------------------------------------------------
__device__ __forceinline__ v16bf load_frag(const bf16* __restrict__ tile, int rowOff, int lane) {
  const bf16* p = tile + (rowOff + (lane & 15)) * LDSS + ((lane >> 4) << 3);
  v16bf f;
  ((uint4*)&f)[0] = ((const uint4*)p)[0];        // k = hi + 0..7
  ((uint4*)&f)[1] = ((const uint4*)(p + 16))[0]; // k = 16 + hi + 0..7
  return f;
}

// ---------------------------------------------------------------------------
// 64x64 C-tile GEMM core: C[m,n] += sum_k A[m,k]*B[n,k], bf16 in, f32 acc.
// 4 waves (2x2), each wave owns a 32x32 sub-tile = 2x2 WMMA accumulators.
// Double-buffered LDS; async prefetch of step k+1 overlaps WMMA of step k.
// sA/sB point to [2][64*LDSS] bf16.
// ---------------------------------------------------------------------------
__device__ __forceinline__ void gemm64(const bf16* __restrict__ A, long ldA,
                                       int mBase, int mMax,
                                       const bf16* __restrict__ Bp, long ldB,
                                       int nBase, int nMax,
                                       int K, v8f (&acc)[2][2],
                                       bf16* sA, bf16* sB) {
  int lane = threadIdx.x & 31;
  int wave = threadIdx.x >> 5;
  int wm = (wave >> 1) << 5;
  int wn = (wave & 1) << 5;
  int NK = K >> 5;

  issue_tile_async(A,  ldA, mBase, mMax, 0, sA);
  issue_tile_async(Bp, ldB, nBase, nMax, 0, sB);

  for (int kt = 0; kt < NK; ++kt) {
    bf16* curA = sA + (kt & 1) * TILE_ELEMS;
    bf16* curB = sB + (kt & 1) * TILE_ELEMS;
    wait_async0();        // my async writes to cur buffers are done
    __syncthreads();      // everyone's writes visible; prev-buffer readers done
    if (kt + 1 < NK) {    // prefetch next K-slice into the other buffer
      int nb = (kt + 1) & 1;
      issue_tile_async(A,  ldA, mBase, mMax, (kt + 1) << 5, sA + nb * TILE_ELEMS);
      issue_tile_async(Bp, ldB, nBase, nMax, (kt + 1) << 5, sB + nb * TILE_ELEMS);
    }
#pragma unroll
    for (int tm = 0; tm < 2; ++tm) {
      v16bf fa = load_frag(curA, wm + tm * 16, lane);
#pragma unroll
      for (int tn = 0; tn < 2; ++tn) {
        v16bf fb = load_frag(curB, wn + tn * 16, lane);
        acc[tm][tn] = __builtin_amdgcn_wmma_f32_16x16x32_bf16(
            false, fa, false, fb, (short)0, acc[tm][tn], false, false);
      }
    }
  }
}

#define ACC_INIT()                                   \
  v8f acc[2][2];                                     \
  { const v8f vz = {0.f,0.f,0.f,0.f,0.f,0.f,0.f,0.f};\
    acc[0][0]=vz; acc[0][1]=vz; acc[1][0]=vz; acc[1][1]=vz; }

#define SHARED_TILES()                               \
  __shared__ __align__(16) bf16 sA[2 * TILE_ELEMS];  \
  __shared__ __align__(16) bf16 sB[2 * TILE_ELEMS];

// C-tile epilogue coordinates: m_local = r + (lane>=16 ? 8 : 0), n_local = lane&15
#define EPILOGUE_FOR(mBaseE, nBaseE, BODY)                       \
  { int lane = threadIdx.x & 31, wave = threadIdx.x >> 5;        \
    int wm = (wave >> 1) << 5, wn = (wave & 1) << 5;             \
    _Pragma("unroll") for (int tm = 0; tm < 2; ++tm)             \
    _Pragma("unroll") for (int tn = 0; tn < 2; ++tn)             \
    _Pragma("unroll") for (int r = 0; r < 8; ++r) {              \
      int m = (mBaseE) + wm + tm*16 + ((lane >> 4) << 3) + r;    \
      int n = (nBaseE) + wn + tn*16 + (lane & 15);               \
      float v = acc[tm][tn][r];                                  \
      BODY                                                       \
    } }

// ---------------------------------------------------------------------------
// Stage 0: segment-mean merge + fp32->bf16 convert.
// heads_bf[b, y, h]: y<256 -> hidden; y>=256 -> deps (== xa without ones col).
// ---------------------------------------------------------------------------
__global__ void __launch_bounds__(256)
k_merge(const float* __restrict__ hidden, const int* __restrict__ word_starts,
        bf16* __restrict__ heads_bf) {
  int row = blockIdx.x;               // 0 .. B*512-1
  int b = row >> 9, y = row & 511;
  const float* hsrc = hidden + (long)b * S_ * H_;
  bf16* dst = heads_bf + (long)row * H_;
  if (y < S_) {
    const float* s = hsrc + (long)y * H_;
    for (int h = threadIdx.x; h < H_; h += blockDim.x) dst[h] = (bf16)s[h];
  } else {
    int j = y - S_;
    const int* wst = word_starts + b * S_;
    __shared__ int nvalid;
    if (threadIdx.x == 0) {
      int c = 0;
      for (int t = 0; t < S_; ++t) c += (wst[t] != 0);
      nvalid = c;
    }
    __syncthreads();
    int start = wst[j];
    int end   = (j + 1 < S_) ? wst[j + 1] : 0;
    bool valid = (j + 1) < nvalid;
    int len = end - start; if (len < 1) len = 1;
    float inv = 1.0f / (float)len;
    for (int h = threadIdx.x; h < H_; h += blockDim.x) {
      float sum = 0.f;
      if (valid)
        for (int t = start; t < end; ++t) sum += hsrc[(long)t * H_ + h];
      dst[h] = (bf16)(valid ? sum * inv : 0.f);
    }
  }
}

// W_arc [769,768] -> Warc_t[j,i]=W_arc[i,j] bf16, i<768 (bias row i=768 stays fp32)
__global__ void __launch_bounds__(256)
k_conv_warc(const float* __restrict__ W_arc, bf16* __restrict__ Warc_t) {
  long idx = (long)blockIdx.x * 256 + threadIdx.x;
  if (idx >= (long)H_ * H_) return;
  int j = (int)(idx / H_), i = (int)(idx % H_);
  Warc_t[idx] = (bf16)W_arc[(long)i * H_ + j];
}

// W_rel [L,769,769] -> Wrel_t[o][j,i]=W_rel[o,i,j] bf16, j<769, i<768
__global__ void __launch_bounds__(256)
k_conv_wrel(const float* __restrict__ W_rel, bf16* __restrict__ Wrel_t) {
  long idx = (long)blockIdx.x * 256 + threadIdx.x;
  long total = (long)L_ * HP1 * H_;
  if (idx >= total) return;
  long o = idx / ((long)HP1 * H_);
  long rem = idx % ((long)HP1 * H_);
  int j = (int)(rem / H_), i = (int)(rem % H_);
  Wrel_t[idx] = (bf16)W_rel[((long)o * HP1 + i) * HP1 + j];
}

// ---------------------------------------------------------------------------
// Arc GEMM 1: U[b,x,j] = sum_i xa[x,i]*W_arc[i,j] + W_arc[768,j]  (bf16 out)
// grid (4, 12, B), block 128
// ---------------------------------------------------------------------------
__global__ void __launch_bounds__(128)
k_arcU(const bf16* __restrict__ heads_bf, const bf16* __restrict__ Warc_t,
       const float* __restrict__ W_arc, bf16* __restrict__ U) {
  SHARED_TILES();
  int mt = blockIdx.x, nt = blockIdx.y, b = blockIdx.z;
  ACC_INIT();
  const bf16* A = heads_bf + ((long)b * Y_ + S_) * H_;   // xa rows (deps)
  gemm64(A, H_, mt * 64, S_, Warc_t, H_, nt * 64, H_, H_, acc, sA, sB);
  const float* bias = W_arc + (long)H_ * H_;             // row i=768
  bf16* Ub = U + (long)b * S_ * H_;
  EPILOGUE_FOR(mt * 64, nt * 64, {
    if (m < S_ && n < H_) Ub[(long)m * H_ + n] = (bf16)(v + bias[n]);
  })
}

// Arc GEMM 2: arc_preds[b,x,y] = sum_j U[b,x,j]*heads[b,y,j]
// grid (4, 8, B), block 128
__global__ void __launch_bounds__(128)
k_arcPred(const bf16* __restrict__ U, const bf16* __restrict__ heads_bf,
          float* __restrict__ arc_out) {
  SHARED_TILES();
  int mt = blockIdx.x, nt = blockIdx.y, b = blockIdx.z;
  ACC_INIT();
  const bf16* A  = U + (long)b * S_ * H_;
  const bf16* Bp = heads_bf + (long)b * Y_ * H_;
  gemm64(A, H_, mt * 64, S_, Bp, H_, nt * 64, Y_, H_, acc, sA, sB);
  float* outb = arc_out + (long)b * S_ * Y_;
  EPILOGUE_FOR(mt * 64, nt * 64, {
    if (m < S_ && n < Y_) outb[(long)m * Y_ + n] = v;
  })
}

// ---------------------------------------------------------------------------
// Rel phase A (per b): T[x,o,j] = sum_i xa[x,i]*W_rel[o,i,j] + W_rel[o,768,j]
// grid (4, 13, 37), block 128.  j range 0..768 inclusive (ones col of yr).
// ---------------------------------------------------------------------------
__global__ void __launch_bounds__(128)
k_relA(const bf16* __restrict__ heads_bf, const bf16* __restrict__ Wrel_t,
       const float* __restrict__ W_rel, bf16* __restrict__ Tbuf, int b) {
  SHARED_TILES();
  int mt = blockIdx.x, nt = blockIdx.y, o = blockIdx.z;
  ACC_INIT();
  const bf16* A  = heads_bf + ((long)b * Y_ + S_) * H_;  // xa rows
  const bf16* Bp = Wrel_t + (long)o * HP1 * H_;
  gemm64(A, H_, mt * 64, S_, Bp, H_, nt * 64, HP1, H_, acc, sA, sB);
  const float* bias = W_rel + ((long)o * HP1 + H_) * HP1; // row i=768, len 769
  EPILOGUE_FOR(mt * 64, nt * 64, {
    if (m < S_ && n < HP1)
      Tbuf[((long)m * L_ + o) * JPAD + n] = (bf16)(v + bias[n]);
  })
}

// Rel phase B (per b): rel[b,x,y,o] = sum_{j<768} heads[b,y,j]*T[x,o,j] + T[x,o,768]
// grid (8, 1, 256), block 128. Output block [512,37] per (b,x) is contiguous.
__global__ void __launch_bounds__(128)
k_relB(const bf16* __restrict__ heads_bf, const bf16* __restrict__ Tbuf,
       float* __restrict__ rel_out, int b) {
  SHARED_TILES();
  int mt = blockIdx.x, x = blockIdx.z;
  ACC_INIT();
  const bf16* A  = heads_bf + (long)b * Y_ * H_;
  const bf16* Bp = Tbuf + (long)x * L_ * JPAD;
  gemm64(A, H_, mt * 64, Y_, Bp, JPAD, 0, L_, H_, acc, sA, sB);
  float* outx = rel_out + ((long)b * S_ + x) * Y_ * L_;
  EPILOGUE_FOR(mt * 64, 0, {
    if (m < Y_ && n < L_) {
      float biasv = (float)Bp[(long)n * JPAD + H_];      // T[x,o,768]
      outx[(long)m * L_ + n] = v + biasv;
    }
  })
}

// ---------------------------------------------------------------------------
// Loss: per (b,x) log-softmax over 512 arc scores + 37 rel scores; masked sums.
// ---------------------------------------------------------------------------
__global__ void __launch_bounds__(256)
k_loss(const float* __restrict__ arc, const float* __restrict__ rel,
       const int* __restrict__ labels_arcs, const int* __restrict__ labels_rels,
       float* __restrict__ accum) {
  int bx = blockIdx.x;
  int b = bx >> 8, x = bx & 255;
  __shared__ float red[256];
  int t = threadIdx.x;
  int la = labels_arcs[b * S_ + x];
  bool mask = (la != 0);                         // PAD == 0
  int safe = la < 0 ? 0 : (la > Y_ - 1 ? Y_ - 1 : la);
  const float* row = arc + ((long)b * S_ + x) * Y_;

  float mv = -3.0e38f;
  for (int i = t; i < Y_; i += 256) mv = fmaxf(mv, row[i]);
  red[t] = mv; __syncthreads();
  for (int s = 128; s > 0; s >>= 1) { if (t < s) red[t] = fmaxf(red[t], red[t + s]); __syncthreads(); }
  float mx = red[0]; __syncthreads();
  float sm = 0.f;
  for (int i = t; i < Y_; i += 256) sm += __expf(row[i] - mx);
  red[t] = sm; __syncthreads();
  for (int s = 128; s > 0; s >>= 1) { if (t < s) red[t] += red[t + s]; __syncthreads(); }
  float lse = mx + __logf(red[0]); __syncthreads();

  const float* rrow = rel + (((long)b * S_ + x) * Y_ + safe) * L_;
  float rv = (t < L_) ? rrow[t] : -3.0e38f;
  red[t] = rv; __syncthreads();
  for (int s = 128; s > 0; s >>= 1) { if (t < s) red[t] = fmaxf(red[t], red[t + s]); __syncthreads(); }
  float rmx = red[0]; __syncthreads();
  red[t] = (t < L_) ? __expf(rv - rmx) : 0.f; __syncthreads();
  for (int s = 128; s > 0; s >>= 1) { if (t < s) red[t] += red[t + s]; __syncthreads(); }
  float rlse = rmx + __logf(red[0]);

  if (t == 0 && mask) {
    int lr = labels_rels[b * S_ + x];
    if (lr < 0) lr = 0; if (lr > L_ - 1) lr = L_ - 1;
    atomicAdd(&accum[0], -(row[safe] - lse));
    atomicAdd(&accum[1], -(rrow[lr] - rlse));
    atomicAdd(&accum[2], 1.0f);
  }
}

__global__ void k_zero(float* accum) { if (threadIdx.x < 16) accum[threadIdx.x] = 0.f; }

__global__ void k_final(const float* __restrict__ accum, float* __restrict__ out_loss) {
  float n = accum[2];
  out_loss[0] = (n > 0.f) ? (accum[0] + accum[1]) / n : 0.f;
}

// ---------------------------------------------------------------------------
extern "C" void kernel_launch(void* const* d_in, const int* in_sizes, int n_in,
                              void* d_out, int out_size, void* d_ws, size_t ws_size,
                              hipStream_t stream) {
  (void)in_sizes; (void)n_in; (void)out_size;
  const float* hidden      = (const float*)d_in[0];
  const float* W_arc       = (const float*)d_in[1];
  const float* W_rel       = (const float*)d_in[2];
  const int*   word_starts = (const int*)d_in[3];
  const int*   labels_arcs = (const int*)d_in[4];
  const int*   labels_rels = (const int*)d_in[5];

  float* out   = (float*)d_out;
  float* d_loss = out;
  float* d_rel  = out + 1;                                  // [B,S,2S,L]
  float* d_arc  = out + 1 + (long)B_ * S_ * Y_ * L_;        // [B,S,2S]

  char* ws = (char*)d_ws;
  size_t off = 0;
  auto take = [&](size_t bytes) {
    char* p = ws + off;
    off = (off + bytes + 255) & ~(size_t)255;
    return p;
  };
  bf16*  heads_bf = (bf16*)take((size_t)B_ * Y_ * H_ * 2);        // 12.6 MB
  bf16*  Warc_t   = (bf16*)take((size_t)H_ * H_ * 2);             //  1.2 MB
  bf16*  U        = (bf16*)take((size_t)B_ * S_ * H_ * 2);        //  6.3 MB
  bf16*  Wrel_t   = (bf16*)take((size_t)L_ * HP1 * H_ * 2);       // 43.7 MB
  bf16*  Tbuf     = (bf16*)take((size_t)S_ * L_ * JPAD * 2);      // 14.7 MB (per-b reuse)
  float* accum    = (float*)take(64);
  if (off > ws_size) return;  // workspace too small (deterministic no-op)

  k_zero<<<1, 16, 0, stream>>>(accum);
  k_merge<<<dim3(B_ * Y_), dim3(256), 0, stream>>>(hidden, word_starts, heads_bf);
  k_conv_warc<<<dim3((H_ * H_ + 255) / 256), dim3(256), 0, stream>>>(W_arc, Warc_t);
  {
    long total = (long)L_ * HP1 * H_;
    k_conv_wrel<<<dim3((unsigned)((total + 255) / 256)), dim3(256), 0, stream>>>(W_rel, Wrel_t);
  }
  k_arcU<<<dim3(4, 12, B_), dim3(128), 0, stream>>>(heads_bf, Warc_t, W_arc, U);
  k_arcPred<<<dim3(4, 8, B_), dim3(128), 0, stream>>>(U, heads_bf, d_arc);
  for (int b = 0; b < B_; ++b) {
    k_relA<<<dim3(4, 13, L_), dim3(128), 0, stream>>>(heads_bf, Wrel_t, W_rel, Tbuf, b);
    k_relB<<<dim3(8, 1, S_), dim3(128), 0, stream>>>(heads_bf, Tbuf, d_rel, b);
  }
  k_loss<<<dim3(B_ * S_), dim3(256), 0, stream>>>(d_arc, d_rel, labels_arcs, labels_rels, accum);
  k_final<<<1, 1, 0, stream>>>(accum, d_loss);
}
